// Down_SelfAttention2D_44178033607174
// MI455X (gfx1250) — matrix operand order repaired and checked by reading, not compiled, not run on hardware
//
#include <hip/hip_runtime.h>
#include <hip/hip_bf16.h>
#include <cstdint>
#include <cstddef>

// ---------------------------------------------------------------------------
// Geometry (fixed by the reference): B=4, Cin=64, H=W=256, k=s=4
//   -> patches N = 64*64 = 4096, K = Cin*4*4 = 1024, O = 128
// ---------------------------------------------------------------------------

typedef __attribute__((ext_vector_type(16))) __bf16 v16bf;
typedef __attribute__((ext_vector_type(8)))  __bf16 v8bf;
typedef __attribute__((ext_vector_type(8)))  float  v8f;

#define WMMA_BF16(A, B, C) \
  __builtin_amdgcn_wmma_f32_16x16x32_bf16(false, (A), false, (B), (short)0, (C), false, false)

// ---------------------------------------------------------------------------
// Stage 1: fp32 OIHW weights -> bf16 [mat][o][k] (flat OIHW == patch K order)
// ---------------------------------------------------------------------------
__global__ __launch_bounds__(256) void cvt_weights(const float* __restrict__ Wq,
                                                   const float* __restrict__ Wk,
                                                   const float* __restrict__ Wv,
                                                   __bf16* __restrict__ W2) {
  const int idx = blockIdx.x * 256 + threadIdx.x;        // 0 .. 3*131072-1 (grid exact)
  const float* src = (idx < 131072) ? Wq : (idx < 262144 ? Wk : Wv);
  const int i = idx & (131072 - 1);
  W2[idx] = (__bf16)src[i];
}

// ---------------------------------------------------------------------------
// Stage 2: QKV patch-GEMM.  One wave computes a 16(n) x 128(o) tile for one of
// {q,k,v}.  A (16x32 bf16) is built from x on the fly; B is a contiguous v16bf
// load from the bf16 weights.  Output: Qb/Kb row-major [b][n][o] bf16,
// V transposed Vt [b][o][n] bf16 (so attention B-tiles are contiguous).
// ---------------------------------------------------------------------------
__global__ __launch_bounds__(128) void qkv_gemm(const float* __restrict__ x,
                                                const __bf16* __restrict__ W2,
                                                const float* __restrict__ bq,
                                                const float* __restrict__ bk,
                                                const float* __restrict__ bv,
                                                __bf16* __restrict__ Qb,
                                                __bf16* __restrict__ Kb,
                                                __bf16* __restrict__ Vt) {
  const int wave = threadIdx.x >> 5;
  const int lane = threadIdx.x & 31;
  const int h  = lane >> 4;      // lane half (A/B layout)
  const int cl = lane & 15;      // A row / B column within tile
  const int mat = blockIdx.y;    // 0=q 1=k 2=v
  const int b   = blockIdx.z;
  const int ntile = blockIdx.x * 4 + wave;

  const int n  = ntile * 16 + cl;          // this lane's A-matrix row (patch id)
  const int ny = n >> 6, nx = n & 63;

  const __bf16* Wm  = W2 + (size_t)mat * 131072;
  const float* bias = (mat == 0) ? bq : (mat == 1 ? bk : bv);
  const float* xb0  = x + (size_t)b * (64 * 256 * 256);

  const v8f vzero = {0.f, 0.f, 0.f, 0.f, 0.f, 0.f, 0.f, 0.f};
  v8f acc[8];
#pragma unroll
  for (int t = 0; t < 8; ++t) acc[t] = vzero;

  for (int kc = 0; kc < 32; ++kc) {         // K = 1024 in chunks of 32
    // --- A tile: lane (h,cl) holds K = kc*32 + 16h + {0..15}
    v16bf a;
    {
      const int k0 = kc * 32 + 16 * h;      // first group of 8 (elems 0..7)
      const int cA = k0 >> 4;
      const int kyA = (k0 & 15) >> 2;       // always 0 (k0 is a multiple of 16)
      const float* p0 = xb0 + (((size_t)cA * 256 + (ny * 4 + kyA)) << 8) + nx * 4;
      const float4 f0 = *(const float4*)p0;
      const float4 f1 = *(const float4*)(p0 + 256);
      a[0] = (__bf16)f0.x; a[1] = (__bf16)f0.y; a[2] = (__bf16)f0.z; a[3] = (__bf16)f0.w;
      a[4] = (__bf16)f1.x; a[5] = (__bf16)f1.y; a[6] = (__bf16)f1.z; a[7] = (__bf16)f1.w;

      const int k1 = k0 + 8;                // second group of 8 (elems 8..15)
      const int cB = k1 >> 4;
      const int kyB = (k1 & 15) >> 2;
      const float* p1 = xb0 + (((size_t)cB * 256 + (ny * 4 + kyB)) << 8) + nx * 4;
      const float4 g0 = *(const float4*)p1;
      const float4 g1 = *(const float4*)(p1 + 256);
      a[8]  = (__bf16)g0.x; a[9]  = (__bf16)g0.y; a[10] = (__bf16)g0.z; a[11] = (__bf16)g0.w;
      a[12] = (__bf16)g1.x; a[13] = (__bf16)g1.y; a[14] = (__bf16)g1.z; a[15] = (__bf16)g1.w;
    }

    // --- B tiles: lane (h,cl) holds column o = t*16+cl, K = kc*32 + 16h + e
#pragma unroll
    for (int t = 0; t < 8; ++t) {
      const int o = t * 16 + cl;
      const v16bf bw = *(const v16bf*)(Wm + (size_t)o * 1024 + kc * 32 + 16 * h);
      acc[t] = WMMA_BF16(a, bw, acc[t]);
    }
  }

  // Epilogue: + bias, bf16 store.  D layout: row n = v+8h, col o = t*16+cl.
#pragma unroll
  for (int t = 0; t < 8; ++t) {
    const int o = t * 16 + cl;
    const float bf = bias[o];
#pragma unroll
    for (int v = 0; v < 8; ++v) {
      const int nn = ntile * 16 + v + 8 * h;
      const float val = acc[t][v] + bf;
      if (mat == 0)      Qb[((size_t)(b * 4096 + nn) << 7) + o] = (__bf16)val;
      else if (mat == 1) Kb[((size_t)(b * 4096 + nn) << 7) + o] = (__bf16)val;
      else               Vt[(((size_t)b * 128 + o) << 12) + nn] = (__bf16)val;
    }
  }
}

// ---------------------------------------------------------------------------
// Stage 3: flash attention.  One wave owns 16 query rows x 128 channels.
// Streams keys 32 at a time: 8 WMMAs for S (contraction over 128 channels),
// online softmax (shfl_xor butterflies within 16-lane halves), P goes through
// LDS for the D-layout -> A-layout transpose, then 8 WMMAs accumulate P*V.
// ---------------------------------------------------------------------------
__global__ __launch_bounds__(128) void attn(const __bf16* __restrict__ Qb,
                                            const __bf16* __restrict__ Kb,
                                            const __bf16* __restrict__ Vt,
                                            float* __restrict__ out) {
  __shared__ __bf16 lds[4 * 512];          // 4 waves x (16x32 bf16) P staging
  const int wave = threadIdx.x >> 5;
  const int lane = threadIdx.x & 31;
  const int h  = lane >> 4;
  const int cl = lane & 15;
  const int b  = blockIdx.y;
  const int q0 = (blockIdx.x * 4 + wave) * 16;
  __bf16* pl = lds + wave * 512;

  // Q as 4 A-tiles (16 q-rows x K=32-channel chunks), kept in registers.
  v16bf qa[4];
  {
    const __bf16* qrow = Qb + ((size_t)(b * 4096 + q0 + cl) << 7);
#pragma unroll
    for (int ck = 0; ck < 4; ++ck) {
      const v8bf lo = *(const v8bf*)(qrow + ck * 32 + 16 * h);
      const v8bf hi = *(const v8bf*)(qrow + ck * 32 + 16 * h + 8);
#pragma unroll
      for (int e = 0; e < 8; ++e) { qa[ck][e] = lo[e]; qa[ck][8 + e] = hi[e]; }
    }
  }

  const v8f vzero = {0.f, 0.f, 0.f, 0.f, 0.f, 0.f, 0.f, 0.f};
  float mrow[8], lrow[8], alpha[8];
  v8f acc[8];
#pragma unroll
  for (int v = 0; v < 8; ++v) { mrow[v] = -3.0e38f; lrow[v] = 0.f; }
#pragma unroll
  for (int t = 0; t < 8; ++t) acc[t] = vzero;

  for (int m0 = 0; m0 < 4096; m0 += 32) {
    // ---- S = Q K^T for two 16-key subtiles (D rows = q = v+8h, cols = key)
    v8f s0 = vzero, s1 = vzero;
    const __bf16* kr0 = Kb + ((size_t)(b * 4096 + m0 + cl) << 7) + 16 * h;
    const __bf16* kr1 = kr0 + (16 << 7);
#pragma unroll
    for (int ck = 0; ck < 4; ++ck) {
      const v16bf kb0 = *(const v16bf*)(kr0 + ck * 32);
      const v16bf kb1 = *(const v16bf*)(kr1 + ck * 32);
      s0 = WMMA_BF16(qa[ck], kb0, s0);
      s1 = WMMA_BF16(qa[ck], kb1, s1);
    }

    // ---- online softmax over this 32-key slab (fp32)
#pragma unroll
    for (int v = 0; v < 8; ++v) {
      float rm = fmaxf(s0[v], s1[v]);
      rm = fmaxf(rm, __shfl_xor(rm, 1, 32));
      rm = fmaxf(rm, __shfl_xor(rm, 2, 32));
      rm = fmaxf(rm, __shfl_xor(rm, 4, 32));
      rm = fmaxf(rm, __shfl_xor(rm, 8, 32));
      const float mn = fmaxf(mrow[v], rm);
      const float al = __expf(mrow[v] - mn);
      const float p0 = __expf(s0[v] - mn);
      const float p1 = __expf(s1[v] - mn);
      s0[v] = p0; s1[v] = p1;
      float rs = p0 + p1;
      rs += __shfl_xor(rs, 1, 32);
      rs += __shfl_xor(rs, 2, 32);
      rs += __shfl_xor(rs, 4, 32);
      rs += __shfl_xor(rs, 8, 32);
      lrow[v] = lrow[v] * al + rs;
      mrow[v] = mn;
      alpha[v] = al;
    }
#pragma unroll
    for (int t = 0; t < 8; ++t)
#pragma unroll
      for (int v = 0; v < 8; ++v) acc[t][v] *= alpha[v];

    // ---- P: D-layout -> bf16 16x32 tile in LDS -> reload in A layout
#pragma unroll
    for (int v = 0; v < 8; ++v) {
      pl[(v + 8 * h) * 32 + cl]      = (__bf16)s0[v];
      pl[(v + 8 * h) * 32 + 16 + cl] = (__bf16)s1[v];
    }
    asm volatile("s_wait_dscnt 0" ::: "memory");  // per-wave DS store->load order
    v16bf pa;
    {
      const __bf16* pr = pl + cl * 32;
      const v8bf lo = *(const v8bf*)(pr + 16 * h);
      const v8bf hi = *(const v8bf*)(pr + 16 * h + 8);
#pragma unroll
      for (int e = 0; e < 8; ++e) { pa[e] = lo[e]; pa[8 + e] = hi[e]; }
    }

    if (m0 + 32 < 4096)  // warm L2/L0 for the next key slab
      __builtin_prefetch(Kb + ((size_t)(b * 4096 + m0 + 32 + cl) << 7), 0, 1);

    // ---- O += P (16x32) * V (32 keys x 16 chans) per channel tile
#pragma unroll
    for (int t = 0; t < 8; ++t) {
      const v16bf vb =
          *(const v16bf*)(Vt + (((size_t)b * 128 + t * 16 + cl) << 12) + m0 + 16 * h);
      acc[t] = WMMA_BF16(pa, vb, acc[t]);
    }
  }

  // Epilogue: out[b][o][n] = acc / l  (D rows n = q0+v+8h, cols o = t*16+cl)
#pragma unroll
  for (int t = 0; t < 8; ++t) {
    const int o = t * 16 + cl;
#pragma unroll
    for (int v = 0; v < 8; ++v) {
      const int n = q0 + v + 8 * h;
      out[(((size_t)b * 128 + o) << 12) + n] = acc[t][v] / lrow[v];
    }
  }
}

// ---------------------------------------------------------------------------
// Launch: d_in order = x, Wq, bq, Wk, bk, Wv, bv (all fp32).  Output fp32
// [4][128][64][64].  Workspace: Qb(4MiB) | Kb(4MiB) | Vt(4MiB) | W2(768KiB).
// ---------------------------------------------------------------------------
extern "C" void kernel_launch(void* const* d_in, const int* in_sizes, int n_in,
                              void* d_out, int out_size, void* d_ws, size_t ws_size,
                              hipStream_t stream) {
  (void)in_sizes; (void)n_in; (void)out_size; (void)ws_size;
  const float* x  = (const float*)d_in[0];
  const float* Wq = (const float*)d_in[1];
  const float* bq = (const float*)d_in[2];
  const float* Wk = (const float*)d_in[3];
  const float* bk = (const float*)d_in[4];
  const float* Wv = (const float*)d_in[5];
  const float* bv = (const float*)d_in[6];
  float* out = (float*)d_out;

  char* ws = (char*)d_ws;
  __bf16* Qb = (__bf16*)(ws);
  __bf16* Kb = (__bf16*)(ws + (size_t)4 * 1024 * 1024);
  __bf16* Vt = (__bf16*)(ws + (size_t)8 * 1024 * 1024);
  __bf16* W2 = (__bf16*)(ws + (size_t)12 * 1024 * 1024);

  cvt_weights<<<1536, 256, 0, stream>>>(Wq, Wk, Wv, W2);                 // 3*131072 elems
  qkv_gemm<<<dim3(64, 3, 4), 128, 0, stream>>>(x, W2, bq, bk, bv, Qb, Kb, Vt);
  attn<<<dim3(64, 4), 128, 0, stream>>>(Qb, Kb, Vt, out);
}
